// Seq2SeqLSTM_27848567947306
// MI455X (gfx1250) — matrix operand order repaired and checked by reading, not compiled, not run on hardware
//
#include <hip/hip_runtime.h>
#include <cstdint>
#include <cstddef>

typedef __attribute__((ext_vector_type(16))) _Float16 v16h;
typedef __attribute__((ext_vector_type(8)))  _Float16 v8h;
typedef __attribute__((ext_vector_type(8)))  float    v8f;
typedef __attribute__((ext_vector_type(4)))  unsigned int u32x4;
typedef __attribute__((ext_vector_type(4)))  int      i32x4;
typedef __attribute__((ext_vector_type(8)))  int      i32x8;

#define B_   64
#define T_   512
#define H_   1024
#define GATES_ (4 * H_)

#ifndef __has_builtin
#define __has_builtin(x) 0
#endif
#if defined(__HIP_DEVICE_COMPILE__) && __has_builtin(__builtin_amdgcn_tensor_load_to_lds)
#define USE_TDM 1
#else
#define USE_TDM 0
#endif

// ---------------------------------------------------------------------------
// WMMA helpers (CDNA5 16x16x32 f16 -> f32), layouts per ISA 7.12.2
// ---------------------------------------------------------------------------
__device__ __forceinline__ v8f wmma16(v16h a, v16h b, v8f c) {
  // (neg_a, A, neg_b, B, c_mod, C, reuse_a, reuse_b)
  return __builtin_amdgcn_wmma_f32_16x16x32_f16(false, a, false, b, (short)0, c,
                                                false, false);
}

// A fragment (16x32, f16) from a row-major matrix: lane m=lane&15,
// halves j=0..7 -> K = k0 + hiOff + j ; j=8..15 -> K = k0 + 16 + hiOff + (j-8)
// hiOff = 8 for lanes 16..31. Two contiguous 16-byte loads per lane.
__device__ __forceinline__ v16h load_a_frag(const _Float16* A, int ld, int m0,
                                            int k0, int lane) {
  const int m  = m0 + (lane & 15);
  const int hi = (lane & 16) ? 8 : 0;
  const _Float16* p = A + (size_t)m * ld + k0 + hi;
  union { v16h v; v8h h[2]; } u;
  u.h[0] = *(const v8h*)(p);
  u.h[1] = *(const v8h*)(p + 16);
  return u.v;
}

// B fragment (32x16, f16) from pre-swizzled storage: tile = 1024 B,
// each lane's 16 halves contiguous -> single 32-byte vector load.
__device__ __forceinline__ v16h load_b_frag(const _Float16* Wsw, int tile, int lane) {
  return *(const v16h*)(Wsw + (size_t)tile * 512 + lane * 16);
}

__device__ __forceinline__ float sigmoid_f(float x) { return 1.0f / (1.0f + __expf(-x)); }

// ---------------------------------------------------------------------------
// Tensor Data Mover: 2-D tile load global->LDS (D# per ISA cdna5 8.3/8.4).
// tile_d0 in data_size units (8B here), tile_d1 rows, row stride in units.
// ---------------------------------------------------------------------------
#if USE_TDM
__device__ __forceinline__ void tdm_load_2d(unsigned lds_off, const void* gptr,
                                            unsigned tile_d0, unsigned tile_d1,
                                            unsigned long long stride_units,
                                            unsigned data_size_code) {
  const unsigned long long ga = (unsigned long long)(uintptr_t)gptr;
  u32x4 g0;
  g0[0] = 1u;                                        // count=1, user desc
  g0[1] = lds_off;                                   // lds_addr (bytes)
  g0[2] = (unsigned)(ga & 0xFFFFFFFFu);              // global_addr lo
  g0[3] = (unsigned)((ga >> 32) & 0x01FFFFFFu) | 0x80000000u;  // addr hi | type=2
  i32x8 g1;
  g1[0] = (int)(data_size_code << 16);               // wg_mask=0, data_size
  g1[1] = (int)((tile_d0 & 0xFFFFu) << 16);          // tensor_dim0[15:0] (=tile_d0)
  g1[2] = (int)((tile_d0 >> 16) | ((tile_d1 & 0xFFFFu) << 16)); // dim0 hi | tensor_dim1 lo
  g1[3] = (int)((tile_d1 >> 16) | ((tile_d0 & 0xFFFFu) << 16)); // dim1 hi | tile_dim0
  g1[4] = (int)(tile_d1 & 0xFFFFu);                  // tile_dim1, tile_dim2=0
  g1[5] = (int)(stride_units & 0xFFFFFFFFull);       // tensor_dim0_stride lo
  g1[6] = (int)((stride_units >> 32) & 0xFFFFull);   // stride hi, dim1_stride=0
  g1[7] = 0;
  i32x4 g2 = {0, 0, 0, 0};
  i32x4 g3 = {0, 0, 0, 0};
#if __has_include(<hip/amd_detail/amd_gfx1250_TDM.h>)
  i32x8 g4 = {0, 0, 0, 0, 0, 0, 0, 0};
  __builtin_amdgcn_tensor_load_to_lds(g0, g1, g2, g3, g4, 0);
#else
  __builtin_amdgcn_tensor_load_to_lds(g0, g1, g2, g3, 0);
#endif
}
#endif

// ---------------------------------------------------------------------------
// Device-wide barrier (generation/sense counter in global scratch)
// ---------------------------------------------------------------------------
__device__ __forceinline__ void grid_sync(unsigned* bar, unsigned nwg) {
  __threadfence();   // every wave drains its stores to device scope
  __syncthreads();
  if (threadIdx.x == 0) {
    unsigned g0 = __hip_atomic_load(bar + 1, __ATOMIC_ACQUIRE, __HIP_MEMORY_SCOPE_AGENT);
    unsigned arrived =
        __hip_atomic_fetch_add(bar, 1u, __ATOMIC_ACQ_REL, __HIP_MEMORY_SCOPE_AGENT);
    if (arrived == nwg - 1) {
      __hip_atomic_store(bar, 0u, __ATOMIC_RELAXED, __HIP_MEMORY_SCOPE_AGENT);
      __hip_atomic_fetch_add(bar + 1, 1u, __ATOMIC_ACQ_REL, __HIP_MEMORY_SCOPE_AGENT);
    } else {
      while (__hip_atomic_load(bar + 1, __ATOMIC_ACQUIRE, __HIP_MEMORY_SCOPE_AGENT) == g0)
        __builtin_amdgcn_s_sleep(2);
    }
  }
  __syncthreads();
}

// ---------------------------------------------------------------------------
// Setup kernels: bias combine + zero init, f32->f16 transpose of inputs,
// f32->f16 swizzle of weights into B-fragment tile order.
// ---------------------------------------------------------------------------
__global__ void setup_misc(const float* __restrict__ ebih, const float* __restrict__ ebhh,
                           const float* __restrict__ dbih, const float* __restrict__ dbhh,
                           float* __restrict__ encB, float* __restrict__ decB,
                           _Float16* __restrict__ h0, unsigned* __restrict__ bar) {
  const int i = blockIdx.x * 256 + threadIdx.x;
  if (i < GATES_) { encB[i] = ebih[i] + ebhh[i]; decB[i] = dbih[i] + dbhh[i]; }
  if (i < B_ * H_) h0[i] = (_Float16)0.0f;
  if (i < 2) bar[i] = 0u;
}

// out[t][b][k] = (f16) X[b][t][k]
__global__ void convert_x(const float* __restrict__ X, _Float16* __restrict__ out,
                          int B, int T, int K) {
  const long long o = (long long)blockIdx.x * 256 + threadIdx.x;
  if (o >= (long long)B * T * K) return;
  const int k = (int)(o % K);
  const long long r = o / K;
  const int b = (int)(r % B);
  const int t = (int)(r / B);
  out[o] = (_Float16)X[((size_t)b * T + t) * K + k];
}

// W[N][K] f32 -> swizzled f16 tiles: tile (nt,kt) at (nt*KT+kt)*512 halves,
// element (lane,j): n = nt*16 + lane%16 ; k = kt*32 + (lane<16?0:8) + (j&7) + (j<8?0:16)
__global__ void convert_swizzle_w(const float* __restrict__ W, _Float16* __restrict__ out,
                                  int N, int K) {
  const int KT = K >> 5;
  const long long o = (long long)blockIdx.x * 256 + threadIdx.x;
  if (o >= (long long)N * K) return;
  const int tile = (int)(o >> 9);
  const int rem  = (int)(o & 511);
  const int nt = tile / KT, kt = tile % KT;
  const int lanei = rem >> 4, j = rem & 15;
  const int n = nt * 16 + (lanei & 15);
  const int k = kt * 32 + ((lanei & 16) ? 8 : 0) + (j & 7) + ((j & 8) ? 16 : 0);
  out[o] = (_Float16)W[(size_t)n * K + k];
}

// ---------------------------------------------------------------------------
// Persistent LSTM: 64 WGs, WG wg owns h columns [wg*16, wg*16+16) and the
// 4 corresponding 16-row gate blocks (i,f,g,o). Whh slice (64x1024 f16,
// 128 KB) staged in LDS once via TDM; Wih slice streamed from L2. One
// device-wide barrier per timestep; x_{t+1} prefetched during activations.
// ---------------------------------------------------------------------------
template <int KX, bool IS_DEC>
__global__ __launch_bounds__(256, 1)
void lstm_seq_kernel(const _Float16* __restrict__ xseq,   // [T][B][KX] f16
                     const _Float16* __restrict__ wWih,   // swizzled, 256*(KX/32) tiles
                     const _Float16* __restrict__ wWhh,   // swizzled, 256*32 tiles
                     const float*    __restrict__ bias,   // [4H] combined
                     _Float16* __restrict__ hPing,        // enc ping
                     _Float16* __restrict__ hPong,        // enc pong
                     const _Float16* __restrict__ hInit,  // dec: encoder final h
                     float*    __restrict__ cState,       // [B][H] handoff
                     _Float16* __restrict__ decHs,        // dec: [T][B][H]
                     unsigned* __restrict__ bar) {
  __shared__ _Float16 sWhh[64 * H_];   // 128 KB: WG's Whh slice (swizzled tiles)
  __shared__ float    sGates[64 * 64]; // 16 KB : [batch][i|f|g|o x16]
  __shared__ float    sC[64 * 16];     // 4 KB  : cell state slice
  __shared__ float    sB[64];

  const int wg   = blockIdx.x;
  const int tid  = threadIdx.x;
  const int wave = tid >> 5, lane = tid & 31;
  const int ln    = wave & 3;   // local gate group 0..3 (i,f,g,o)
  const int mhalf = wave >> 2;  // batch half
  constexpr int KTX = KX / 32;

  // Stage Whh slice: gate-row tiles nt = {g*64+wg : g=0..3}, all 32 k-tiles.
  // In swizzled storage these are 4 contiguous 32 KB runs, 2 MB apart ->
  // one TDM 2-D tile: 4 rows x 4096 (8B units), row stride 262144 units.
#if USE_TDM
  if (wave == 0) {
    tdm_load_2d((unsigned)(uintptr_t)(void*)sWhh,
                wWhh + (size_t)wg * 32 * 512,
                /*tile_d0=*/4096u, /*tile_d1=*/4u,
                /*stride=*/262144ull, /*8B units=*/3u);
    __builtin_amdgcn_s_wait_tensorcnt(0);
  }
#else
  for (int g = 0; g < 4; ++g) {
    const v8h* src = (const v8h*)(wWhh + (size_t)(g * 64 + wg) * 32 * 512);
    v8h* dst = (v8h*)(sWhh + g * 32 * 512);
    for (int i = tid; i < (32 * 512) / 8; i += 256) dst[i] = src[i];
  }
#endif
  if (tid < 64) sB[tid] = bias[(tid >> 4) * H_ + wg * 16 + (tid & 15)];
  for (int i = tid; i < 64 * 16; i += 256)
    sC[i] = IS_DEC ? cState[(size_t)(i >> 4) * H_ + wg * 16 + (i & 15)] : 0.0f;
  __syncthreads();

  const int m0a = mhalf * 32;
  const int m0b = mhalf * 32 + 16;

  for (int t = 0; t < T_; ++t) {
    const _Float16* hprev;
    if constexpr (IS_DEC)
      hprev = (t == 0) ? hInit : (decHs + (size_t)(t - 1) * B_ * H_);
    else
      hprev = (t & 1) ? hPong : hPing;
    const _Float16* xt = xseq + (size_t)t * B_ * KX;

    v8f acc0 = {}; v8f acc1 = {};
    const int ntg = ln * 64 + wg;  // global 16-row gate tile index

    // gates += x_t @ Wih^T   (B fragments streamed from L2)
    for (int kt = 0; kt < KTX; ++kt) {
      v16h bf = load_b_frag(wWih, ntg * KTX + kt, lane);
      v16h a0 = load_a_frag(xt, KX, m0a, kt * 32, lane);
      v16h a1 = load_a_frag(xt, KX, m0b, kt * 32, lane);
      acc0 = wmma16(a0, bf, acc0);
      acc1 = wmma16(a1, bf, acc1);
    }
    // gates += h_{t-1} @ Whh^T   (weights LDS-resident)
    for (int kt = 0; kt < 32; ++kt) {
      v16h bf = load_b_frag(sWhh, ln * 32 + kt, lane);
      v16h a0 = load_a_frag(hprev, H_, m0a, kt * 32, lane);
      v16h a1 = load_a_frag(hprev, H_, m0b, kt * 32, lane);
      acc0 = wmma16(a0, bf, acc0);
      acc1 = wmma16(a1, bf, acc1);
    }

    // Scatter C fragments to LDS (C layout: row = r + 8*(lane>=16), col = lane%16)
    {
      const int col = lane & 15;
      const int rb  = (lane & 16) ? 8 : 0;
#pragma unroll
      for (int r = 0; r < 8; ++r) {
        sGates[(m0a + rb + r) * 64 + ln * 16 + col] = acc0[r];
        sGates[(m0b + rb + r) * 64 + ln * 16 + col] = acc1[r];
      }
    }

    // Prefetch next timestep's x tile while activations run.
    if (t + 1 < T_) {
      const char* nx = (const char*)(xseq + (size_t)(t + 1) * B_ * KX);
      for (int o = tid * 256; o < B_ * KX * 2; o += 256 * 256)
        __builtin_prefetch(nx + o, 0, 1);
    }
    __syncthreads();

    _Float16* hout;
    if constexpr (IS_DEC) hout = decHs + (size_t)t * B_ * H_;
    else                  hout = (t & 1) ? hPing : hPong;

    for (int i = tid; i < 64 * 16; i += 256) {
      const int mb = i >> 4, col = i & 15;
      const float gi = sGates[mb * 64 +      col] + sB[     col];
      const float gf = sGates[mb * 64 + 16 + col] + sB[16 + col];
      const float gg = sGates[mb * 64 + 32 + col] + sB[32 + col];
      const float go = sGates[mb * 64 + 48 + col] + sB[48 + col];
      const float c  = sigmoid_f(gf) * sC[i] + sigmoid_f(gi) * tanhf(gg);
      sC[i] = c;
      hout[(size_t)mb * H_ + wg * 16 + col] = (_Float16)(sigmoid_f(go) * tanhf(c));
    }
    grid_sync(bar, gridDim.x);
  }

  if constexpr (!IS_DEC) {
    for (int i = tid; i < 64 * 16; i += 256)
      cState[(size_t)(i >> 4) * H_ + wg * 16 + (i & 15)] = sC[i];
  }
}

// ---------------------------------------------------------------------------
// Fused MLP: relu(X@W1^T+b1) -> relu(@W2^T+b2) -> @W3^T+b3.
// One WG per timestep (64 rows = full batch). Intermediates in LDS.
// ---------------------------------------------------------------------------
__global__ __launch_bounds__(256, 1)
void mlp_kernel(const _Float16* __restrict__ X,   // decHs: [T][B][H] = rows t*64+b
                const _Float16* __restrict__ w1, const float* __restrict__ b1,
                const _Float16* __restrict__ w2, const float* __restrict__ b2,
                const _Float16* __restrict__ w3, const float* __restrict__ b3,
                float* __restrict__ out) {       // [B][T][256]
  __shared__ _Float16 sX1[64 * 256];  // 32 KB
  __shared__ _Float16 sX2[64 * 128];  // 16 KB

  const int t    = blockIdx.x;
  const int tid  = threadIdx.x;
  const int wave = tid >> 5, lane = tid & 31;
  const _Float16* Xwg = X + (size_t)t * 64 * H_;

  // ---- layer 1: [64x1024] @ [1024x256] ----
  {
    v8f acc[8] = {};
    for (int kt = 0; kt < 32; ++kt) {
      v16h bf0 = load_b_frag(w1, (wave * 2 + 0) * 32 + kt, lane);
      v16h bf1 = load_b_frag(w1, (wave * 2 + 1) * 32 + kt, lane);
#pragma unroll
      for (int mt = 0; mt < 4; ++mt) {
        v16h a = load_a_frag(Xwg, H_, mt * 16, kt * 32, lane);
        acc[mt * 2 + 0] = wmma16(a, bf0, acc[mt * 2 + 0]);
        acc[mt * 2 + 1] = wmma16(a, bf1, acc[mt * 2 + 1]);
      }
    }
    const int col = lane & 15, rb = (lane & 16) ? 8 : 0;
    for (int mt = 0; mt < 4; ++mt)
      for (int j = 0; j < 2; ++j) {
        const int nc = (wave * 2 + j) * 16 + col;
        const float bb = b1[nc];
#pragma unroll
        for (int r = 0; r < 8; ++r)
          sX1[(mt * 16 + rb + r) * 256 + nc] = (_Float16)fmaxf(acc[mt * 2 + j][r] + bb, 0.0f);
      }
  }
  __syncthreads();

  // ---- layer 2: [64x256] @ [256x128] ----
  {
    v8f acc[4] = {};
    for (int kt = 0; kt < 8; ++kt) {
      v16h bf = load_b_frag(w2, wave * 8 + kt, lane);
#pragma unroll
      for (int mt = 0; mt < 4; ++mt) {
        v16h a = load_a_frag(sX1, 256, mt * 16, kt * 32, lane);
        acc[mt] = wmma16(a, bf, acc[mt]);
      }
    }
    const int col = lane & 15, rb = (lane & 16) ? 8 : 0;
    const int nc = wave * 16 + col;
    const float bb = b2[nc];
    for (int mt = 0; mt < 4; ++mt)
#pragma unroll
      for (int r = 0; r < 8; ++r)
        sX2[(mt * 16 + rb + r) * 128 + nc] = (_Float16)fmaxf(acc[mt][r] + bb, 0.0f);
  }
  __syncthreads();

  // ---- layer 3: [64x128] @ [128x256] -> out f32 ----
  {
    v8f acc[8] = {};
    for (int kt = 0; kt < 4; ++kt) {
      v16h bf0 = load_b_frag(w3, (wave * 2 + 0) * 4 + kt, lane);
      v16h bf1 = load_b_frag(w3, (wave * 2 + 1) * 4 + kt, lane);
#pragma unroll
      for (int mt = 0; mt < 4; ++mt) {
        v16h a = load_a_frag(sX2, 128, mt * 16, kt * 32, lane);
        acc[mt * 2 + 0] = wmma16(a, bf0, acc[mt * 2 + 0]);
        acc[mt * 2 + 1] = wmma16(a, bf1, acc[mt * 2 + 1]);
      }
    }
    const int col = lane & 15, rb = (lane & 16) ? 8 : 0;
    for (int mt = 0; mt < 4; ++mt)
      for (int j = 0; j < 2; ++j) {
        const int nc = (wave * 2 + j) * 16 + col;
        const float bb = b3[nc];
#pragma unroll
        for (int r = 0; r < 8; ++r) {
          const int brow = mt * 16 + rb + r;  // batch index (rows within t)
          out[((size_t)brow * T_ + t) * 256 + nc] = acc[mt * 2 + j][r] + bb;
        }
      }
  }
}

// ---------------------------------------------------------------------------
// Host launcher
// ---------------------------------------------------------------------------
extern "C" void kernel_launch(void* const* d_in, const int* in_sizes, int n_in,
                              void* d_out, int out_size, void* d_ws, size_t ws_size,
                              hipStream_t stream) {
  (void)in_sizes; (void)n_in; (void)out_size; (void)ws_size;
  const float* in_seq  = (const float*)d_in[0];
  const float* tgt_seq = (const float*)d_in[1];
  const float* eWih = (const float*)d_in[2];
  const float* eWhh = (const float*)d_in[3];
  const float* ebih = (const float*)d_in[4];
  const float* ebhh = (const float*)d_in[5];
  const float* dWih = (const float*)d_in[6];
  const float* dWhh = (const float*)d_in[7];
  const float* dbih = (const float*)d_in[8];
  const float* dbhh = (const float*)d_in[9];
  const float* W1 = (const float*)d_in[10];
  const float* b1 = (const float*)d_in[11];
  const float* W2 = (const float*)d_in[12];
  const float* b2 = (const float*)d_in[13];
  const float* W3 = (const float*)d_in[14];
  const float* b3 = (const float*)d_in[15];
  float* out = (float*)d_out;

  char* ws = (char*)d_ws;
  size_t off = 0;
  auto take = [&](size_t bytes) -> char* {
    char* p = ws + off;
    off = (off + bytes + 255) & ~(size_t)255;
    return p;
  };
  unsigned*  bar    = (unsigned*)take(256);
  float*     encB   = (float*)take((size_t)GATES_ * 4);
  float*     decB   = (float*)take((size_t)GATES_ * 4);
  _Float16*  h0     = (_Float16*)take((size_t)B_ * H_ * 2);
  _Float16*  h1     = (_Float16*)take((size_t)B_ * H_ * 2);
  float*     cInit  = (float*)take((size_t)B_ * H_ * 4);
  _Float16*  xenc   = (_Float16*)take((size_t)T_ * B_ * 256 * 2);
  _Float16*  xdec   = (_Float16*)take((size_t)T_ * B_ * H_ * 2);
  _Float16*  decHs  = (_Float16*)take((size_t)T_ * B_ * H_ * 2);
  _Float16*  swEWih = (_Float16*)take((size_t)GATES_ * 256 * 2);
  _Float16*  swEWhh = (_Float16*)take((size_t)GATES_ * H_ * 2);
  _Float16*  swDWih = (_Float16*)take((size_t)GATES_ * H_ * 2);
  _Float16*  swDWhh = (_Float16*)take((size_t)GATES_ * H_ * 2);
  _Float16*  swW1   = (_Float16*)take((size_t)256 * 1024 * 2);
  _Float16*  swW2   = (_Float16*)take((size_t)128 * 256 * 2);
  _Float16*  swW3   = (_Float16*)take((size_t)256 * 128 * 2);

  setup_misc<<<(B_ * H_) / 256, 256, 0, stream>>>(ebih, ebhh, dbih, dbhh, encB, decB, h0, bar);
  convert_x<<<(B_ * T_ * 256) / 256, 256, 0, stream>>>(in_seq, xenc, B_, T_, 256);
  convert_x<<<(B_ * T_ * H_) / 256, 256, 0, stream>>>(tgt_seq, xdec, B_, T_, H_);
  convert_swizzle_w<<<(GATES_ * 256) / 256, 256, 0, stream>>>(eWih, swEWih, GATES_, 256);
  convert_swizzle_w<<<(GATES_ * H_) / 256, 256, 0, stream>>>(eWhh, swEWhh, GATES_, H_);
  convert_swizzle_w<<<(GATES_ * H_) / 256, 256, 0, stream>>>(dWih, swDWih, GATES_, H_);
  convert_swizzle_w<<<(GATES_ * H_) / 256, 256, 0, stream>>>(dWhh, swDWhh, GATES_, H_);
  convert_swizzle_w<<<(256 * 1024) / 256, 256, 0, stream>>>(W1, swW1, 256, 1024);
  convert_swizzle_w<<<(128 * 256) / 256, 256, 0, stream>>>(W2, swW2, 128, 256);
  convert_swizzle_w<<<(256 * 128) / 256, 256, 0, stream>>>(W3, swW3, 256, 128);

  // Encoder: KX=256, ping-pong h buffers; final h lands in h0, c -> cInit.
  lstm_seq_kernel<256, false><<<64, 256, 0, stream>>>(
      xenc, swEWih, swEWhh, encB, h0, h1, (const _Float16*)nullptr, cInit,
      (_Float16*)nullptr, bar);
  // Decoder: KX=1024, h history written straight into decHs.
  lstm_seq_kernel<1024, true><<<64, 256, 0, stream>>>(
      xdec, swDWih, swDWhh, decB, (_Float16*)nullptr, (_Float16*)nullptr, h0,
      cInit, decHs, bar);
  // Fused 3-layer MLP over all [B*T] rows.
  mlp_kernel<<<T_, 256, 0, stream>>>(decHs, swW1, b1, swW2, b2, swW3, b3, out);
}
